// AttnBlockv3_24077586661592
// MI455X (gfx1250) — matrix operand order repaired and checked by reading, not compiled, not run on hardware
//
#include <hip/hip_runtime.h>
#include <hip/hip_bf16.h>

typedef __attribute__((ext_vector_type(16))) _Float16 v16h;
typedef __attribute__((ext_vector_type(8)))  _Float16 v8h;
typedef __attribute__((ext_vector_type(8)))  float    v8f;
typedef int v4i __attribute__((vector_size(16)));

static constexpr int Bc   = 16;            // batch
static constexpr int Cc   = 128;           // channels
static constexpr int Nsp  = 64 * 64;       // spatial positions per batch (4096)
static constexpr int Mrows = Bc * Nsp;     // 65536 pixel rows
static constexpr int NG  = 32;             // groups
static constexpr int CPG = 4;              // channels per group

#define WMMA_F32_F16(a, b, c) \
  __builtin_amdgcn_wmma_f32_16x16x32_f16(false, (a), false, (b), (short)0, (c), false, false)

__device__ __forceinline__ v16h cat16(v8h lo, v8h hi) {
  return __builtin_shufflevector(lo, hi, 0,1,2,3,4,5,6,7,8,9,10,11,12,13,14,15);
}

// ---------------------------------------------------------------------------
// Async global->LDS staging (gfx1250 GLOBAL_LOAD_ASYNC_TO_LDS_B128).
// Signature (from clang diagnostic): void(v4i AS1*, v4i AS3*, imm int, imm int)
// ---------------------------------------------------------------------------
#if defined(__has_builtin)
#if __has_builtin(__builtin_amdgcn_global_load_async_to_lds_b128) && \
    __has_builtin(__builtin_amdgcn_s_wait_asynccnt)
#define HAVE_ASYNC_LDS 1
#endif
#endif
#ifndef HAVE_ASYNC_LDS
#define HAVE_ASYNC_LDS 0
#endif

typedef __attribute__((address_space(1))) v4i GAv4i;
typedef __attribute__((address_space(3))) v4i LAv4i;

__device__ __forceinline__ void cp_b128(const _Float16* g, _Float16* l) {
#if HAVE_ASYNC_LDS
  __builtin_amdgcn_global_load_async_to_lds_b128((GAv4i*)g, (LAv4i*)l, 0, 0);
#else
  *(v8h*)l = *(const v8h*)g;
#endif
}

// ---------------------------------------------------------------------------
// Convert the four 128x128 f32 weight matrices to f16, TRANSPOSED ([n][k]),
// so GEMM B-fragments are K-contiguous (b128 LDS loads, no scalar gathers).
// ---------------------------------------------------------------------------
__global__ __launch_bounds__(256) void cvt_w(const float* __restrict__ wq,
                                             const float* __restrict__ wk,
                                             const float* __restrict__ wv,
                                             const float* __restrict__ wo,
                                             _Float16* __restrict__ out) {
  const int i = blockIdx.x * 256 + threadIdx.x;            // 0 .. 65535
  const int m = i >> 14, rem = i & 16383;
  const int n = rem >> 7, k = rem & 127;
  const float* s = (m == 0) ? wq : (m == 1) ? wk : (m == 2) ? wv : wo;
  out[i] = (_Float16)s[k * Cc + n];                        // out[m][n][k] = w[k][n]
}

// ---------------------------------------------------------------------------
// GroupNorm: one block per (batch, group). Writes normalized f16 [B, N, C].
// ---------------------------------------------------------------------------
__global__ __launch_bounds__(256) void gnorm(const float* __restrict__ x,
                                             const float* __restrict__ gs,
                                             const float* __restrict__ gb,
                                             _Float16* __restrict__ h) {
  const int b = blockIdx.x, g = blockIdx.y;
  const int c0 = g * CPG;
  const float* xb = x + (size_t)b * Nsp * Cc;
  const int total = Nsp * CPG;                             // 16384
  float sum = 0.f, sq = 0.f;
  for (int i = threadIdx.x; i < total; i += 256) {
    const int n = i >> 2, c = c0 + (i & 3);
    const float v = xb[(size_t)n * Cc + c];
    sum += v; sq += v * v;
  }
  __shared__ float r0[256], r1[256];
  r0[threadIdx.x] = sum; r1[threadIdx.x] = sq;
  __syncthreads();
  for (int st = 128; st > 0; st >>= 1) {
    if ((int)threadIdx.x < st) {
      r0[threadIdx.x] += r0[threadIdx.x + st];
      r1[threadIdx.x] += r1[threadIdx.x + st];
    }
    __syncthreads();
  }
  const float mean = r0[0] / (float)total;
  const float var  = r1[0] / (float)total - mean * mean;
  const float inv  = rsqrtf(var + 1e-6f);
  _Float16* hb = h + (size_t)b * Nsp * Cc;
  for (int i = threadIdx.x; i < total; i += 256) {
    const int n = i >> 2, c = c0 + (i & 3);
    const float v = xb[(size_t)n * Cc + c];
    hb[(size_t)n * Cc + c] = (_Float16)((v - mean) * inv * gs[c] + gb[c]);
  }
}

// ---------------------------------------------------------------------------
// Transpose f16 [B, N, C] -> [B, C, N] (for the P@V operand of attention).
// ---------------------------------------------------------------------------
__global__ __launch_bounds__(256) void transpose_cn(const _Float16* __restrict__ src,
                                                    _Float16* __restrict__ dst) {
  __shared__ _Float16 t[32][33];
  const int b = blockIdx.z;
  const int n0 = blockIdx.x * 32, c0 = blockIdx.y * 32;
  const int tid = threadIdx.x;
  const int r = tid >> 3, q = (tid & 7) * 4;
  const _Float16* s = src + ((size_t)b * Nsp + n0) * Cc + c0;
#pragma unroll
  for (int i = 0; i < 4; ++i) t[r][q + i] = s[(size_t)r * Cc + q + i];
  __syncthreads();
  _Float16* d = dst + ((size_t)b * Cc + c0) * Nsp + n0;
#pragma unroll
  for (int i = 0; i < 4; ++i) d[(size_t)r * Nsp + q + i] = t[q + i][r];
}

// ---------------------------------------------------------------------------
// GEMM: out[M,128] = A[M,128] @ W[128,128] + bias, with Wt = W^T staged in LDS.
// B-fragment loads are software-pipelined one WMMA ahead.
// mode 0: store f16.  mode 1: out_f32 = resid + result (final projection).
// ---------------------------------------------------------------------------
__global__ __launch_bounds__(256) void gemm128(const _Float16* __restrict__ A,
                                               const _Float16* __restrict__ Wt,
                                               const float* __restrict__ bias,
                                               _Float16* __restrict__ outH,
                                               const float* __restrict__ resid,
                                               float* __restrict__ outF,
                                               int mode) {
  __shared__ __align__(32) _Float16 wl[128 * 128];         // 32 KB, [n][k]
  for (int i = threadIdx.x; i < 2048; i += 256)
    cp_b128(Wt + i * 8, wl + i * 8);
#if HAVE_ASYNC_LDS
  __builtin_amdgcn_s_wait_asynccnt(0);
#endif
  __syncthreads();

  const int wave = threadIdx.x >> 5, lane = threadIdx.x & 31;
  const size_t row0 = (size_t)blockIdx.x * 128 + wave * 16;
  const _Float16* Ab = A + row0 * Cc;
  const int ar  = lane & 15, akb = (lane >> 4) * 16;       // A-frag row / K base
  const int bc  = lane & 15, bkb = (lane >> 4) * 16;       // B-frag col / K base

  v16h a[4];
#pragma unroll
  for (int s = 0; s < 4; ++s) {
    v8h lo = *(const v8h*)(Ab + ar * Cc + s * 32 + akb);
    v8h hi = *(const v8h*)(Ab + ar * Cc + s * 32 + akb + 8);
    a[s] = cat16(lo, hi);
  }

  v8f acc[8] = {};
  v16h bcur = *(const v16h*)(wl + bc * 128 + bkb);         // (s=0, nt=0)
#pragma unroll
  for (int s = 0; s < 4; ++s) {
#pragma unroll
    for (int nt = 0; nt < 8; ++nt) {
      v16h bnext = bcur;
      const int idx = s * 8 + nt;
      if (idx < 31) {
        const int s2 = (idx + 1) >> 3, nt2 = (idx + 1) & 7;
        bnext = *(const v16h*)(wl + (nt2 * 16 + bc) * 128 + s2 * 32 + bkb);
      }
      acc[nt] = WMMA_F32_F16(a[s], bcur, acc[nt]);
      bcur = bnext;
    }
  }

  const int rb = (lane >> 4) * 8;                          // C-frag row base
#pragma unroll
  for (int nt = 0; nt < 8; ++nt) {
    const int n0 = nt * 16;
    const float bv = bias[n0 + bc];
#pragma unroll
    for (int r = 0; r < 8; ++r) {
      const size_t idx = (row0 + rb + r) * Cc + n0 + bc;
      const float v = acc[nt][r] + bv;
      if (mode == 0) outH[idx] = (_Float16)v;
      else           outF[idx] = resid[idx] + v;
    }
  }
}

// ---------------------------------------------------------------------------
// Flash attention: grid (N/128, B); 8 waves/block, each wave owns 16 query
// rows. 32 keys per step; K tile (row-major) and V^T tile (channel-major)
// DOUBLE-BUFFERED in LDS via async copies so DMA overlaps compute, and all
// WMMA operand loads are pipelined one matrix op ahead.
// ---------------------------------------------------------------------------
__global__ __launch_bounds__(256) void fattn(const _Float16* __restrict__ Q,
                                             const _Float16* __restrict__ K,
                                             const _Float16* __restrict__ Vt,
                                             _Float16* __restrict__ O) {
  __shared__ __align__(32) _Float16 kl[2][32 * 128];       // 2 x 8 KB: [key][c]
  __shared__ __align__(32) _Float16 vl[2][32 * 128];       // 2 x 8 KB: [c][key]
  __shared__ __align__(32) _Float16 pl[8 * 512];           // 8 KB: P bounce / wave

  const int b = blockIdx.y;
  const int wave = threadIdx.x >> 5, lane = threadIdx.x & 31;
  const size_t base = (size_t)b * Nsp * Cc;
  const size_t q0 = (size_t)blockIdx.x * 128 + wave * 16;
  const int ar  = lane & 15, akb = (lane >> 4) * 16;
  const int bc  = lane & 15, bkb = (lane >> 4) * 16;
  const int rb  = (lane >> 4) * 8;

  // Stage 32-key K and V^T tiles (4 async b128 copies per thread per tile).
  auto stage = [&](int jt) {
    const int j = jt * 32;
    const _Float16* Kg = K  + base + (size_t)j * Cc;       // 32 keys x 128 ch
    const _Float16* Vg = Vt + base + j;                    // 128 ch x 32 keys
    _Float16* kd = kl[jt & 1];
    _Float16* vd = vl[jt & 1];
    for (int i = threadIdx.x; i < 1024; i += 256) {
      if (i < 512) {
        const int key = i >> 4, part = i & 15;
        cp_b128(Kg + (size_t)key * Cc + part * 8, kd + key * 128 + part * 8);
      } else {
        const int ii = i - 512, c = ii >> 2, part = ii & 3;
        cp_b128(Vg + (size_t)c * Nsp + part * 8, vd + c * 32 + part * 8);
      }
    }
  };

  v16h qf[4];
  const _Float16* Qb = Q + base + q0 * Cc;
#pragma unroll
  for (int s = 0; s < 4; ++s) {
    v8h lo = *(const v8h*)(Qb + ar * Cc + s * 32 + akb);
    v8h hi = *(const v8h*)(Qb + ar * Cc + s * 32 + akb + 8);
    qf[s] = cat16(lo, hi);
  }

  v8f o[8] = {};
  float m[8], l[8];
#pragma unroll
  for (int r = 0; r < 8; ++r) { m[r] = -1e30f; l[r] = 0.f; }

  _Float16* pw = pl + wave * 512;
  const float sc = 0.08838834764831845f;                   // 1/sqrt(128)
  const int T = Nsp / 32;                                  // 128 key tiles

  stage(0);
  for (int jt = 0; jt < T; ++jt) {
    if (jt + 1 < T) {
      stage(jt + 1);                                       // DMA next tile
#if HAVE_ASYNC_LDS
      __builtin_amdgcn_s_wait_asynccnt(4);                 // tile jt resident
#endif
    } else {
#if HAVE_ASYNC_LDS
      __builtin_amdgcn_s_wait_asynccnt(0);
#endif
    }
    __syncthreads();                                       // visible to all waves
    const _Float16* kc = kl[jt & 1];
    const _Float16* vc = vl[jt & 1];

    // S = Q @ K^T for two 16-key column tiles (loads pipelined ahead).
    v8f s0 = {}, s1 = {};
    {
      v16h k0 = *(const v16h*)(kc + bc * 128 + bkb);
      v16h k1 = *(const v16h*)(kc + (bc + 16) * 128 + bkb);
#pragma unroll
      for (int s = 0; s < 4; ++s) {
        v16h n0 = k0, n1 = k1;
        if (s < 3) {
          n0 = *(const v16h*)(kc + bc * 128 + (s + 1) * 32 + bkb);
          n1 = *(const v16h*)(kc + (bc + 16) * 128 + (s + 1) * 32 + bkb);
        }
        s0 = WMMA_F32_F16(qf[s], k0, s0);
        s1 = WMMA_F32_F16(qf[s], k1, s1);
        k0 = n0; k1 = n1;
      }
    }

    // Online softmax per query row (row's 16 cols live in one 16-lane half).
#pragma unroll
    for (int r = 0; r < 8; ++r) {
      const float v0 = s0[r] * sc, v1 = s1[r] * sc;
      float mx = fmaxf(v0, v1);
      mx = fmaxf(mx, __shfl_xor(mx, 1, 32));
      mx = fmaxf(mx, __shfl_xor(mx, 2, 32));
      mx = fmaxf(mx, __shfl_xor(mx, 4, 32));
      mx = fmaxf(mx, __shfl_xor(mx, 8, 32));
      const float mn = fmaxf(m[r], mx);
      const float p0 = __expf(v0 - mn), p1 = __expf(v1 - mn);
      float rs = p0 + p1;
      rs += __shfl_xor(rs, 1, 32);
      rs += __shfl_xor(rs, 2, 32);
      rs += __shfl_xor(rs, 4, 32);
      rs += __shfl_xor(rs, 8, 32);
      const float al = __expf(m[r] - mn);
      l[r] = l[r] * al + rs;
      m[r] = mn;
      s0[r] = p0; s1[r] = p1;
#pragma unroll
      for (int nt = 0; nt < 8; ++nt) o[nt][r] *= al;
    }

    // Re-layout P (C-frag) -> A-frag via this wave's private LDS slab.
#pragma unroll
    for (int r = 0; r < 8; ++r) {
      pw[(rb + r) * 32 + bc]      = (_Float16)s0[r];
      pw[(rb + r) * 32 + 16 + bc] = (_Float16)s1[r];
    }
    const v8h plo = *(const v8h*)(pw + ar * 32 + akb);
    const v8h phi = *(const v8h*)(pw + ar * 32 + akb + 8);
    const v16h pf = cat16(plo, phi);

    // O += P @ V (K = 32 keys), V^T tile channel-major; loads pipelined.
    {
      v16h vcur = *(const v16h*)(vc + bc * 32 + bkb);      // nt = 0
#pragma unroll
      for (int nt = 0; nt < 8; ++nt) {
        v16h vnext = vcur;
        if (nt < 7)
          vnext = *(const v16h*)(vc + ((nt + 1) * 16 + bc) * 32 + bkb);
        o[nt] = WMMA_F32_F16(pf, vcur, o[nt]);
        vcur = vnext;
      }
    }
    __syncthreads();                                       // buffer reuse guard
  }

  // Normalize and store f16 attention output.
  _Float16* Ob = O + base + q0 * Cc;
#pragma unroll
  for (int nt = 0; nt < 8; ++nt) {
#pragma unroll
    for (int r = 0; r < 8; ++r) {
      const float v = o[nt][r] / l[r];
      Ob[(size_t)(rb + r) * Cc + nt * 16 + bc] = (_Float16)v;
    }
  }
}

// ---------------------------------------------------------------------------
extern "C" void kernel_launch(void* const* d_in, const int* in_sizes, int n_in,
                              void* d_out, int out_size, void* d_ws, size_t ws_size,
                              hipStream_t stream) {
  const float* x  = (const float*)d_in[0];
  const float* gs = (const float*)d_in[1];
  const float* gb = (const float*)d_in[2];
  const float* wq = (const float*)d_in[3];
  const float* bq = (const float*)d_in[4];
  const float* wk = (const float*)d_in[5];
  const float* bk = (const float*)d_in[6];
  const float* wv = (const float*)d_in[7];
  const float* bv = (const float*)d_in[8];
  const float* wo = (const float*)d_in[9];
  const float* bo = (const float*)d_in[10];
  float* out = (float*)d_out;

  char* ws = (char*)d_ws;
  const size_t TEN = (size_t)Mrows * Cc;                   // 8,388,608 elems
  _Float16* hf  = (_Float16*)(ws);                         // 16 MB
  _Float16* qf  = (_Float16*)(ws + TEN * 2);               // 16 MB
  _Float16* kf  = (_Float16*)(ws + TEN * 4);               // 16 MB
  _Float16* vf  = (_Float16*)(ws + TEN * 6);               // 16 MB
  _Float16* of  = (_Float16*)(ws + TEN * 8);               // 16 MB
  _Float16* vt  = (_Float16*)(ws + TEN * 10);              // 16 MB (V^T)
  _Float16* w16 = (_Float16*)(ws + TEN * 12);              // 128 KB (4 x W^T)

  cvt_w<<<256, 256, 0, stream>>>(wq, wk, wv, wo, w16);
  gnorm<<<dim3(Bc, NG), 256, 0, stream>>>(x, gs, gb, hf);

  const int gblocks = Mrows / 128;                         // 512
  gemm128<<<gblocks, 256, 0, stream>>>(hf, w16,         bq, qf, nullptr, nullptr, 0);
  gemm128<<<gblocks, 256, 0, stream>>>(hf, w16 + 16384, bk, kf, nullptr, nullptr, 0);
  gemm128<<<gblocks, 256, 0, stream>>>(hf, w16 + 32768, bv, vf, nullptr, nullptr, 0);

  transpose_cn<<<dim3(Nsp / 32, Cc / 32, Bc), 256, 0, stream>>>(vf, vt);

  fattn<<<dim3(Nsp / 128, Bc), 256, 0, stream>>>(qf, kf, vt, of);

  gemm128<<<gblocks, 256, 0, stream>>>(of, w16 + 49152, bo, nullptr, x, out, 1);
}
